// Net_2327872274495
// MI455X (gfx1250) — matrix operand order repaired and checked by reading, not compiled, not run on hardware
//
#include <hip/hip_runtime.h>
#include <math.h>

#define NN    65536
#define CC    128
#define EE    786432
#define BB    128
#define NPG   512

typedef __bf16 v16bf __attribute__((ext_vector_type(16)));
typedef float  v8f   __attribute__((ext_vector_type(8)));

__device__ __forceinline__ __bf16 f2bf(float f) { return (__bf16)f; }

// ---------------------------------------------------------------------------
// init: mask = 1, readout accumulator = 0
__global__ __launch_bounds__(256) void init_kernel(int* mask, float* hsum) {
    int idx = blockIdx.x * 256 + threadIdx.x;
    if (idx < NN) mask[idx] = 1;
    if (idx < BB * 256) hsum[idx] = 0.f;
}

// zero a float4 buffer
__global__ __launch_bounds__(256) void zero4_kernel(float4* buf, int n4) {
    int idx = blockIdx.x * 256 + threadIdx.x;
    if (idx < n4) buf[idx] = make_float4(0.f, 0.f, 0.f, 0.f);
}

// ---------------------------------------------------------------------------
// weight prep: W is [K][128] f32 (k-major); produce Wt [128][K] bf16 (n-major)
__global__ __launch_bounds__(256) void transpose_bf16_kernel(const float* __restrict__ W,
                                                             unsigned short* __restrict__ WtU,
                                                             int K) {
    int idx = blockIdx.x * 256 + threadIdx.x;
    if (idx >= 128 * K) return;
    int n = idx / K, k = idx % K;
    __bf16 v = f2bf(W[(size_t)k * 128 + n]);
    __bf16* Wt = (__bf16*)WtU;
    Wt[idx] = v;
}

// embedding gather: x[i][:] = emb[ids[i]][:]
__global__ __launch_bounds__(256) void gather_kernel(const int* __restrict__ ids,
                                                     const float* __restrict__ emb,
                                                     float* __restrict__ x) {
    int idx = blockIdx.x * 256 + threadIdx.x;      // over N*32 float4s
    int i = idx >> 5, c4 = idx & 31;
    float4 v = ((const float4*)(emb + (size_t)ids[i] * CC))[c4];
    ((float4*)(x + (size_t)i * CC))[c4] = v;
}

// ---------------------------------------------------------------------------
// WMMA GEMM: out[N,128] = ReLU( [A (k<128) | A2 (k>=128)] @ Wt^T + bias )
// Wt is bf16 [128][KK] (n-major). Weights staged in LDS once per block
// (padded rows for bank rotation); one wave -> 16x128 tile; block = 8 waves.
template <int KK>
__global__ __launch_bounds__(256)
void wmma_gemm_kernel(const float* __restrict__ A, const float* __restrict__ A2,
                      const unsigned short* __restrict__ WtU,
                      const float* __restrict__ bias, float* __restrict__ out) {
    constexpr int LROW = KK + 8;           // +16B pad -> LDS bank rotation
    extern __shared__ char smem[];
    __bf16* wlds = (__bf16*)smem;

    // ---- stage weights: 128 rows of KK bf16, uint4 (16B) granularity ----
    constexpr int SEGS = KK / 8;           // uint4 segments per row
    for (int idx = threadIdx.x; idx < 128 * SEGS; idx += 256) {
        int row = idx / SEGS, seg = idx % SEGS;
        const uint4* srcp = (const uint4*)(WtU + (size_t)row * KK) + seg;
        uint4* dstp = (uint4*)(wlds + (size_t)row * LROW) + seg;
        *dstp = *srcp;
    }
    __syncthreads();

    const int wave = threadIdx.x >> 5;
    const int lane = threadIdx.x & 31;
    const int hl   = lane >> 4;            // which 16-lane half
    const int lr   = lane & 15;
    const int m0   = blockIdx.x * 128 + wave * 16;

    v8f zero = {};
    v8f acc[8];
#pragma unroll
    for (int i = 0; i < 8; ++i) acc[i] = zero;

#pragma unroll
    for (int k0 = 0; k0 < KK; k0 += 32) {
        const float* Asrc = (k0 < 128) ? A : A2;
        const int    kl   = (k0 < 128) ? k0 : (k0 - 128);
        // A fragment: lane(<16): k = kl+0..15 ; lane(>=16): k = kl+16..31
        const float* arow = Asrc + (size_t)(m0 + lr) * CC + kl + hl * 16;
        float4 f0 = *(const float4*)(arow);
        float4 f1 = *(const float4*)(arow + 4);
        float4 f2 = *(const float4*)(arow + 8);
        float4 f3 = *(const float4*)(arow + 12);
        v16bf a;
        a[0]=f2bf(f0.x);  a[1]=f2bf(f0.y);  a[2]=f2bf(f0.z);  a[3]=f2bf(f0.w);
        a[4]=f2bf(f1.x);  a[5]=f2bf(f1.y);  a[6]=f2bf(f1.z);  a[7]=f2bf(f1.w);
        a[8]=f2bf(f2.x);  a[9]=f2bf(f2.y);  a[10]=f2bf(f2.z); a[11]=f2bf(f2.w);
        a[12]=f2bf(f3.x); a[13]=f2bf(f3.y); a[14]=f2bf(f3.z); a[15]=f2bf(f3.w);

        // batch all B fragments (LDS), then the 8 WMMAs
        v16bf bfrag[8];
#pragma unroll
        for (int nt = 0; nt < 8; ++nt)
            bfrag[nt] = *(const v16bf*)(wlds + (size_t)(nt * 16 + lr) * LROW + k0 + hl * 16);
#pragma unroll
        for (int nt = 0; nt < 8; ++nt)
            acc[nt] = __builtin_amdgcn_wmma_f32_16x16x32_bf16(
                false, a, false, bfrag[nt], (short)0, acc[nt], false, false);
    }

    // C/D layout: lane<16 -> N=lane, rows m0+r ; lane>=16 -> N=lane-16, rows m0+8+r
#pragma unroll
    for (int nt = 0; nt < 8; ++nt) {
        const int n = nt * 16 + lr;
        const float bn = bias ? bias[n] : 0.f;
#pragma unroll
        for (int r = 0; r < 8; ++r) {
            int m = m0 + r + hl * 8;
            float v = acc[nt][r] + bn;
            v = v > 0.f ? v : 0.f;
            out[(size_t)m * CC + n] = v;
        }
    }
}

// ---------------------------------------------------------------------------
// per-edge scatter-max: agg[dst] = max(agg[dst], h[src]) for live endpoints.
// h >= 0 (ReLU) and agg zero-initialized, so int-bit atomicMax is exact.
__global__ __launch_bounds__(256)
void edge_max_kernel(const float* __restrict__ h, float* __restrict__ agg,
                     const int* __restrict__ src, const int* __restrict__ dst,
                     const int* __restrict__ mask) {
    int e = blockIdx.x * 2 + (threadIdx.x >> 7);
    int c = threadIdx.x & 127;
    int s = src[e], d = dst[e];
    if (mask[s] & mask[d]) {
        float v = h[(size_t)s * CC + c];
        atomicMax((int*)(agg + (size_t)d * CC + c), __float_as_int(v));
    }
}

// self-loop: agg[i] = max(agg[i], h[i]) for live i (non-atomic, one owner/elem)
__global__ __launch_bounds__(256)
void self_loop_kernel(const float* __restrict__ h, float* __restrict__ agg,
                      const int* __restrict__ mask) {
    int i = blockIdx.x * 2 + (threadIdx.x >> 7);
    int c = threadIdx.x & 127;
    if (mask[i]) {
        size_t o = (size_t)i * CC + c;
        agg[o] = fmaxf(agg[o], h[o]);
    }
}

// ---------------------------------------------------------------------------
// TopK pooling per graph (block = 512 threads, one per node)
__global__ __launch_bounds__(512)
void topk_kernel(const float* __restrict__ y, float* __restrict__ x,
                 int* __restrict__ mask, const float* __restrict__ p, int k) {
    __shared__ float s[NPG];
    __shared__ float f[NPG];
    __shared__ float psq[128];
    int g = blockIdx.x, t = threadIdx.x;
    int i = g * NPG + t;

    if (t < 128) psq[t] = p[t] * p[t];
    __syncthreads();
    for (int off = 64; off >= 1; off >>= 1) {
        if (t < off) psq[t] += psq[t + off];
        __syncthreads();
    }
    float inorm = rsqrtf(psq[0]);

    const float* yr = y + (size_t)i * CC;
    float dot = 0.f;
    for (int c = 0; c < CC; c += 4) {
        float4 v = *(const float4*)(yr + c);
        dot += v.x * p[c] + v.y * p[c + 1] + v.z * p[c + 2] + v.w * p[c + 3];
    }
    float score = dot * inorm;
    int mi = mask[i];
    s[t] = mi ? score : -3.402823e38f;
    __syncthreads();

    // stable rank: count strictly-greater, or equal with smaller index
    float si = s[t];
    int cnt = 0;
    for (int j = 0; j < NPG; ++j) {
        float sj = s[j];
        cnt += (sj > si) || (sj == si && j < t);
    }
    int keep = (cnt < k) && mi;
    mask[i] = keep;
    f[t] = keep ? tanhf(score) : 0.f;
    __syncthreads();

    const float* yb = y + (size_t)g * NPG * CC;
    float* xb = x + (size_t)g * NPG * CC;
    for (int idx = t; idx < NPG * CC; idx += NPG)
        xb[idx] = yb[idx] * f[idx >> 7];
}

// readout accumulate: hsum[g][0:128] += max over live, hsum[g][128:256] += mean
__global__ __launch_bounds__(128)
void readout_kernel(const float* __restrict__ x, const int* __restrict__ mask,
                    float kf, float* __restrict__ hsum) {
    int g = blockIdx.x, c = threadIdx.x;
    const float* xb = x + (size_t)g * NPG * CC;
    const int* mb = mask + g * NPG;
    float mx = -3.402823e38f, sm = 0.f;
    for (int i = 0; i < NPG; ++i) {
        if (mb[i]) {
            float v = xb[(size_t)i * CC + c];
            mx = fmaxf(mx, v);
            sm += v;
        }
    }
    hsum[g * 256 + c] += mx;
    hsum[g * 256 + 128 + c] += sm / kf;
}

// ---------------------------------------------------------------------------
// tiny MLP layer: out[g][n] = act( sum_k in[g][k]*w[k][n] + b[n] )
__global__ __launch_bounds__(256)
void mlp_kernel(const float* __restrict__ in, const float* __restrict__ w,
                const float* __restrict__ b, float* __restrict__ out,
                int rows, int kin, int nout, int act_sigmoid) {
    int idx = blockIdx.x * 256 + threadIdx.x;
    if (idx >= rows * nout) return;
    int g = idx / nout, n = idx % nout;
    float s = b[n];
    for (int k = 0; k < kin; ++k) s += in[(size_t)g * kin + k] * w[(size_t)k * nout + n];
    out[idx] = act_sigmoid ? 1.f / (1.f + expf(-s)) : (s > 0.f ? s : 0.f);
}

// ---------------------------------------------------------------------------
extern "C" void kernel_launch(void* const* d_in, const int* in_sizes, int n_in,
                              void* d_out, int out_size, void* d_ws, size_t ws_size,
                              hipStream_t stream) {
    const int*   node_ids = (const int*)d_in[0];
    const int*   srcE     = (const int*)d_in[1];
    const int*   dstE     = (const int*)d_in[2];
    const float* emb      = (const float*)d_in[3];
    const float* W[3] = {(const float*)d_in[4],  (const float*)d_in[8],  (const float*)d_in[12]};
    const float* bb[3] = {(const float*)d_in[5], (const float*)d_in[9],  (const float*)d_in[13]};
    const float* U[3] = {(const float*)d_in[6],  (const float*)d_in[10], (const float*)d_in[14]};
    const float* pp[3] = {(const float*)d_in[7], (const float*)d_in[11], (const float*)d_in[15]};
    const float* lw1 = (const float*)d_in[16]; const float* lb1 = (const float*)d_in[17];
    const float* lw2 = (const float*)d_in[18]; const float* lb2 = (const float*)d_in[19];
    const float* lw3 = (const float*)d_in[20]; const float* lb3 = (const float*)d_in[21];

    char* ws = (char*)d_ws;
    size_t off = 0;
    auto alloc = [&](size_t bytes) -> void* {
        void* ptr = ws + off;
        off = (off + bytes + 255) & ~(size_t)255;
        return ptr;
    };
    const size_t xcb = (size_t)NN * CC * sizeof(float);
    float* x    = (float*)alloc(xcb);
    float* h    = (float*)alloc(xcb);
    float* agg  = (float*)alloc(xcb);
    float* y    = (float*)alloc(xcb);
    int*   mask = (int*)alloc((size_t)NN * sizeof(int));
    float* hsum = (float*)alloc((size_t)BB * 256 * sizeof(float));
    float* m1   = (float*)alloc((size_t)BB * 128 * sizeof(float));
    float* m2   = (float*)alloc((size_t)BB * 64 * sizeof(float));
    unsigned short* Wt[3];
    unsigned short* Ut[3];
    for (int i = 0; i < 3; ++i) Wt[i] = (unsigned short*)alloc((size_t)128 * 128 * 2);
    for (int i = 0; i < 3; ++i) Ut[i] = (unsigned short*)alloc((size_t)128 * 256 * 2);

    const int KS[3] = {410, 328, 263};
    const size_t lds128 = (size_t)128 * (128 + 8) * 2;   // 34816 B
    const size_t lds256 = (size_t)128 * (256 + 8) * 2;   // 67584 B

    init_kernel<<<NN / 256, 256, 0, stream>>>(mask, hsum);
    for (int i = 0; i < 3; ++i) {
        transpose_bf16_kernel<<<(128 * 128 + 255) / 256, 256, 0, stream>>>(W[i], Wt[i], 128);
        transpose_bf16_kernel<<<(128 * 256 + 255) / 256, 256, 0, stream>>>(U[i], Ut[i], 256);
    }
    gather_kernel<<<(NN * 32) / 256, 256, 0, stream>>>(node_ids, emb, x);

    for (int L = 0; L < 3; ++L) {
        // h = ReLU(x @ W + b)   [WMMA bf16, weights in LDS]
        wmma_gemm_kernel<128><<<NN / 128, 256, lds128, stream>>>(x, nullptr, Wt[L], bb[L], h);
        // agg = 0; scatter-max over edges; self-loop
        zero4_kernel<<<(NN * 32) / 256, 256, 0, stream>>>((float4*)agg, NN * 32);
        edge_max_kernel<<<EE / 2, 256, 0, stream>>>(h, agg, srcE, dstE, mask);
        self_loop_kernel<<<NN / 2, 256, 0, stream>>>(h, agg, mask);
        // y = ReLU(concat(agg, x) @ U)   [WMMA bf16, K=256]
        wmma_gemm_kernel<256><<<NN / 128, 256, lds256, stream>>>(agg, x, Ut[L], nullptr, y);
        // TopK pool: x = y * tanh(score) * keep ; mask updated
        topk_kernel<<<BB, 512, 0, stream>>>(y, x, mask, pp[L], KS[L]);
        // hsum += [gmp || gap]
        readout_kernel<<<BB, 128, 0, stream>>>(x, mask, (float)KS[L], hsum);
    }

    mlp_kernel<<<(BB * 128 + 255) / 256, 256, 0, stream>>>(hsum, lw1, lb1, m1, BB, 256, 128, 0);
    mlp_kernel<<<(BB * 64 + 255) / 256, 256, 0, stream>>>(m1, lw2, lb2, m2, BB, 128, 64, 0);
    mlp_kernel<<<1, 256, 0, stream>>>(m2, lw3, lb3, (float*)d_out, BB, 64, 1, 1);
}
